// Decoder_Stack_59219009077949
// MI455X (gfx1250) — matrix operand order repaired and checked by reading, not compiled, not run on hardware
//
#include <hip/hip_runtime.h>
#include <hip/hip_bf16.h>

typedef unsigned short u16;
typedef unsigned int u32;
typedef unsigned long long u64;
typedef __attribute__((ext_vector_type(16))) __bf16 v16bf;
typedef __attribute__((ext_vector_type(8)))  float  v8f;

// ---------------------------------------------------------------- constants
constexpr int Bb = 2;
constexpr int Ss = 2048;
constexpr int Ee = 1024;
constexpr int Aa = 64;
constexpr int Vv = 32000;
constexpr int Ff = 4 * Ee;       // 4096
constexpr int Mm = Bb * Ss;      // 4096 token rows
constexpr int STACKN = 8;

// ---------------------------------------------------------------- helpers
__device__ __forceinline__ u16 f2bf(float f) {
  u32 u = __float_as_uint(f);
  u32 r = (u + 0x7FFFu + ((u >> 16) & 1u)) >> 16;   // round-to-nearest-even
  return (u16)r;
}

union FragB16 { v16bf v; u32 u[8]; };

__device__ __forceinline__ v8f wmma_bf16(v16bf a, v16bf b, v8f c) {
  // D = A(16x32 bf16) x B(32x16 bf16) + C(16x16 f32)
  return __builtin_amdgcn_wmma_f32_16x16x32_bf16(
      /*neg_a=*/false, a, /*neg_b=*/false, b,
      /*c_mod=*/(short)0, c, /*reuse_a=*/false, /*reuse_b=*/false);
}

// Async 16B/lane copy global -> LDS (ASYNCcnt-tracked DMA; no VGPR data path).
__device__ __forceinline__ void async_copy_b128(const u16* gptr, const u16* lptr) {
  u32 lds = (u32)(size_t)lptr;                     // flat low 32 bits == LDS byte addr
  u64 g   = (u64)(size_t)gptr;
  asm volatile("global_load_async_to_lds_b128 %0, %1, off"
               :: "v"(lds), "v"(g) : "memory");
}
__device__ __forceinline__ void wait_async0() {
  asm volatile("s_wait_asynccnt 0x0" ::: "memory");
}

// ---------------------------------------------------------------- weight prep
// Wt[n*K + k] = (bf16) W[k*N + n]   (transpose so B-fragment k-pairs are contiguous)
__global__ void transpose_to_bf16_kernel(const float* __restrict__ W,
                                         u16* __restrict__ Wt, int K, int N) {
  size_t idx = (size_t)blockIdx.x * blockDim.x + threadIdx.x;
  size_t total = (size_t)K * N;
  if (idx >= total) return;
  int n = (int)(idx / K);
  int k = (int)(idx - (size_t)n * K);
  Wt[idx] = f2bf(W[(size_t)k * N + n]);
}

// Fold the 16 shared heads: WdrT[e*64 + a] = (bf16) sum_h Wd[(h*64+a)*E + e]
__global__ void wd_reduce_kernel(const float* __restrict__ Wd, u16* __restrict__ WdrT) {
  int idx = blockIdx.x * blockDim.x + threadIdx.x;   // over Ee*64
  if (idx >= Ee * Aa) return;
  int e = idx / Aa, a = idx % Aa;
  float s = 0.f;
  #pragma unroll
  for (int h = 0; h < 16; ++h) s += Wd[(size_t)(h * Aa + a) * Ee + e];
  WdrT[idx] = f2bf(s);
}

__global__ void embed_kernel(const int* __restrict__ x, const float* __restrict__ emb,
                             float* __restrict__ h) {
  size_t idx = (size_t)blockIdx.x * blockDim.x + threadIdx.x;   // over Mm*Ee
  if (idx >= (size_t)Mm * Ee) return;
  int i = (int)(idx / Ee), e = (int)(idx % Ee);
  h[idx] = emb[(size_t)x[i] * Ee + e];
}

__global__ void cvt_bf16_kernel(const float* __restrict__ X, u16* __restrict__ Y, size_t n) {
  size_t idx = (size_t)blockIdx.x * blockDim.x + threadIdx.x;
  if (idx < n) Y[idx] = f2bf(X[idx]);
}

// ---------------------------------------------------------------- layernorm -> bf16
__global__ void __launch_bounds__(256)
layernorm_kernel(const float* __restrict__ X, const float* __restrict__ g,
                 const float* __restrict__ be, u16* __restrict__ Z) {
  __shared__ float red[256];
  const int row = blockIdx.x, t = threadIdx.x;
  const float* x = X + (size_t)row * Ee;
  float s = 0.f, s2 = 0.f;
  for (int i = t; i < Ee; i += 256) { float v = x[i]; s += v; s2 += v * v; }
  red[t] = s; __syncthreads();
  for (int off = 128; off; off >>= 1) { if (t < off) red[t] += red[t + off]; __syncthreads(); }
  float mean = red[0] * (1.f / Ee);
  __syncthreads();
  red[t] = s2; __syncthreads();
  for (int off = 128; off; off >>= 1) { if (t < off) red[t] += red[t + off]; __syncthreads(); }
  float var = red[0] * (1.f / Ee) - mean * mean;
  float rstd = rsqrtf(var + 1e-5f);
  u16* z = Z + (size_t)row * Ee;
  for (int i = t; i < Ee; i += 256) z[i] = f2bf((x[i] - mean) * rstd * g[i] + be[i]);
}

// ---------------------------------------------------------------- WMMA GEMM
// C[M,N] = A_bf16[M,K] @ Bt_bf16[N,K]^T (+bias)(+relu)(+resid) -> Cf (f32) / Cb (bf16)
// Block tile 128x64, 8 waves (4x2), each wave 32x32 = 2x2 WMMA tiles, K-step 32.
// Double-buffered: async global->LDS DMA for tile i+1 overlaps WMMA on tile i.
#define GBM 128
#define GBN 64
#define GBK 32
#define LDT 40   // padded LDS row stride (u16), multiple of 8 -> 16B aligned rows

__global__ void __launch_bounds__(256)
gemm_bf16_kernel(const u16* __restrict__ Aq, const u16* __restrict__ Bt,
                 const float* __restrict__ bias, const float* __restrict__ resid,
                 float* __restrict__ Cf, u16* __restrict__ Cb,
                 int M, int N, int K, int relu) {
  __shared__ u16 As[2][GBM * LDT];
  __shared__ u16 Bs[2][GBN * LDT];
  const int t = threadIdx.x;
  const int lane = t & 31, wave = t >> 5;
  const int wm = wave >> 1, wn = wave & 1;
  const int mlane = lane & 15, hi = lane >> 4;
  const int row0 = blockIdx.y * GBM, col0 = blockIdx.x * GBN;

  // per-lane piece of this wave's DMA duty: each async b128 moves 8 rows x 64B
  const int rA = lane >> 2;        // row within an 8-row chunk
  const int cg = lane & 3;         // 16B column group (4 per 32-elem row)

  // Issue this block's k-tile as 24 async b128 DMAs (3 per wave: 2 A-chunks + 1 B-chunk)
  auto issue_tile = [&](int buf, int k0) {
    #pragma unroll
    for (int i = 0; i < 2; ++i) {
      int c   = wave * 2 + i;                     // A chunk 0..15
      int row = c * 8 + rA;                       // 0..127
      async_copy_b128(Aq + (size_t)(row0 + row) * K + k0 + cg * 8,
                      &As[buf][row * LDT + cg * 8]);
    }
    {
      int row = wave * 8 + rA;                    // B chunk row 0..63
      async_copy_b128(Bt + (size_t)(col0 + row) * K + k0 + cg * 8,
                      &Bs[buf][row * LDT + cg * 8]);
    }
  };

  v8f acc[2][2];
  #pragma unroll
  for (int i = 0; i < 2; ++i)
    #pragma unroll
    for (int j = 0; j < 2; ++j) acc[i][j] = (v8f){0,0,0,0,0,0,0,0};

  const int nsteps = K / GBK;
  issue_tile(0, 0);

  for (int s = 0; s < nsteps; ++s) {
    const int cur = s & 1;
    wait_async0();                 // my DMAs for buffer `cur` have landed
    __syncthreads();               // everyone's have
    if (s + 1 < nsteps) issue_tile(cur ^ 1, (s + 1) * GBK);   // overlap with WMMA below

    // build fragments (16-bit A and B layout: VGPR v -> K pair g*16 + hi*8 + 2r)
    FragB16 fa[2], fb[2];
    #pragma unroll
    for (int ti = 0; ti < 2; ++ti) {
      int m = wm * 32 + ti * 16 + mlane;
      #pragma unroll
      for (int vv = 0; vv < 8; ++vv) {
        int kp = (vv >> 2) * 16 + hi * 8 + (vv & 3) * 2;
        fa[ti].u[vv] = *reinterpret_cast<const u32*>(&As[cur][m * LDT + kp]);
      }
    }
    #pragma unroll
    for (int tj = 0; tj < 2; ++tj) {
      int n = wn * 32 + tj * 16 + mlane;
      #pragma unroll
      for (int vv = 0; vv < 8; ++vv) {
        int kp = (vv >> 2) * 16 + hi * 8 + (vv & 3) * 2;
        fb[tj].u[vv] = *reinterpret_cast<const u32*>(&Bs[cur][n * LDT + kp]);
      }
    }
    #pragma unroll
    for (int ti = 0; ti < 2; ++ti)
      #pragma unroll
      for (int tj = 0; tj < 2; ++tj)
        acc[ti][tj] = wmma_bf16(fa[ti].v, fb[tj].v, acc[ti][tj]);

    __syncthreads();               // readers done before next iter's DMA overwrites
  }

  // epilogue (C layout: lanes 0-15: M=v, N=lane; lanes 16-31: M=v+8, N=lane-16)
  #pragma unroll
  for (int ti = 0; ti < 2; ++ti) {
    #pragma unroll
    for (int tj = 0; tj < 2; ++tj) {
      int nn = col0 + wn * 32 + tj * 16 + mlane;
      float bv = bias ? bias[nn] : 0.f;
      #pragma unroll
      for (int vv = 0; vv < 8; ++vv) {
        int mm = row0 + wm * 32 + ti * 16 + vv + 8 * hi;
        float val = acc[ti][tj][vv] + bv;
        if (relu) val = fmaxf(val, 0.f);
        size_t o = (size_t)mm * N + nn;
        if (resid) val += resid[o];
        if (Cf) Cf[o] = val;
        if (Cb) Cb[o] = f2bf(val);
      }
    }
  }
}

// ---------------------------------------------------------------- flash attention
// Per wave: one 16-query tile; A=64 head dim; streams 32-key chunks with online softmax.
__global__ void __launch_bounds__(128)
attention_kernel(const u16* __restrict__ Q, const u16* __restrict__ Km,
                 const u16* __restrict__ Vm, u16* __restrict__ Out) {
  __shared__ u16 Pb[4][16 * 32];                 // per-wave P staging (C -> A relayout)
  const int t = threadIdx.x, lane = t & 31, w = t >> 5;
  const int mlane = lane & 15, hi = lane >> 4;
  const int ntiles = Ss >> 4;
  const int gt = blockIdx.x * 4 + w;
  const int b = gt / ntiles, qt = gt % ntiles;
  const size_t baseQ = ((size_t)b * Ss + (size_t)qt * 16) * Aa;
  const size_t baseK = (size_t)b * Ss * Aa;
  const float scale = 0.125f;                    // ATTN^-0.5

  // Q fragments: 16x64 = two 16x32 A-fragments, pairs contiguous in [row,64]
  FragB16 fq[2];
  #pragma unroll
  for (int g = 0; g < 2; ++g)
    #pragma unroll
    for (int vv = 0; vv < 8; ++vv) {
      int kp = g * 32 + (vv >> 2) * 16 + hi * 8 + (vv & 3) * 2;
      fq[g].u[vv] = *reinterpret_cast<const u32*>(Q + baseQ + (size_t)mlane * Aa + kp);
    }

  v8f o[4];
  #pragma unroll
  for (int i = 0; i < 4; ++i) o[i] = (v8f){0,0,0,0,0,0,0,0};
  float mi[8], li[8];
  #pragma unroll
  for (int i = 0; i < 8; ++i) { mi[i] = -3.0e38f; li[i] = 0.f; }

  for (int kc = 0; kc < Ss; kc += 32) {
    // scores: two 16x16 tiles (keys kc..+15, kc+16..+31), each 2 WMMAs over head dim
    v8f s[2];
    #pragma unroll
    for (int tt = 0; tt < 2; ++tt) {
      FragB16 fk[2];
      #pragma unroll
      for (int g = 0; g < 2; ++g)
        #pragma unroll
        for (int vv = 0; vv < 8; ++vv) {
          int kp = (vv >> 2) * 16 + hi * 8 + (vv & 3) * 2;   // B layout: K pair within chunk
          int key = kc + tt * 16 + mlane;              // N = key
          fk[g].u[vv] = *reinterpret_cast<const u32*>(Km + baseK + (size_t)key * Aa + g * 32 + kp);
        }
      v8f z = (v8f){0,0,0,0,0,0,0,0};
      z = wmma_bf16(fq[0].v, fk[0].v, z);
      s[tt] = wmma_bf16(fq[1].v, fk[1].v, z);
    }

    // online softmax per row (register v == row M = v + 8*hi; keys live across 16 lanes)
    float alpha[8];
    #pragma unroll
    for (int vv = 0; vv < 8; ++vv) {
      float a0 = s[0][vv] * scale, a1 = s[1][vv] * scale;
      float mx = fmaxf(a0, a1);
      #pragma unroll
      for (int off = 8; off >= 1; off >>= 1) mx = fmaxf(mx, __shfl_xor(mx, off, 32));
      float mt = fmaxf(mi[vv], mx);
      float al = __expf(mi[vv] - mt);
      float p0 = __expf(a0 - mt), p1 = __expf(a1 - mt);
      float rs = p0 + p1;
      #pragma unroll
      for (int off = 8; off >= 1; off >>= 1) rs += __shfl_xor(rs, off, 32);
      li[vv] = li[vv] * al + rs;
      mi[vv] = mt;
      alpha[vv] = al;
      int rrow = vv + 8 * hi;
      Pb[w][rrow * 32 + mlane]      = f2bf(p0);
      Pb[w][rrow * 32 + 16 + mlane] = f2bf(p1);
    }
    __syncthreads();   // uniform across block; orders Pb stores vs fragment loads

    // reload P as a 16x32 A-fragment
    FragB16 fp;
    #pragma unroll
    for (int vv = 0; vv < 8; ++vv) {
      int kp = (vv >> 2) * 16 + hi * 8 + (vv & 3) * 2;
      fp.u[vv] = *reinterpret_cast<const u32*>(&Pb[w][mlane * 32 + kp]);
    }

    // rescale O, then O += P @ V_chunk  (4 n-tiles over the 64 head dims)
    #pragma unroll
    for (int tt = 0; tt < 4; ++tt)
      #pragma unroll
      for (int vv = 0; vv < 8; ++vv) o[tt][vv] *= alpha[vv];
    #pragma unroll
    for (int tt = 0; tt < 4; ++tt) {
      FragB16 fv;
      #pragma unroll
      for (int vv = 0; vv < 8; ++vv) {
        int kp = (vv >> 2) * 16 + hi * 8 + (vv & 3) * 2;   // K pair = local key pair
        const u16* p = Vm + baseK + (size_t)(kc + kp) * Aa + tt * 16 + mlane;
        fv.u[vv] = (u32)p[0] | ((u32)p[Aa] << 16);
      }
      o[tt] = wmma_bf16(fp.v, fv.v, o[tt]);
    }
    __syncthreads();
  }

  // normalize + write head output (bf16)
  #pragma unroll
  for (int tt = 0; tt < 4; ++tt)
    #pragma unroll
    for (int vv = 0; vv < 8; ++vv) {
      float val = o[tt][vv] / li[vv];
      int mm = qt * 16 + vv + 8 * hi;
      Out[((size_t)b * Ss + mm) * Aa + tt * 16 + mlane] = f2bf(val);
    }
}

// ---------------------------------------------------------------- launcher
extern "C" void kernel_launch(void* const* d_in, const int* in_sizes, int n_in,
                              void* d_out, int out_size, void* d_ws, size_t ws_size,
                              hipStream_t stream) {
  const int*   x    = (const int*)  d_in[0];
  const float* emb  = (const float*)d_in[1];
  const float* Wq   = (const float*)d_in[2];
  const float* bq   = (const float*)d_in[3];
  const float* Wk   = (const float*)d_in[4];
  const float* bk   = (const float*)d_in[5];
  const float* Wv   = (const float*)d_in[6];
  const float* bv   = (const float*)d_in[7];
  const float* Wd   = (const float*)d_in[8];
  const float* bd   = (const float*)d_in[9];
  const float* g1   = (const float*)d_in[10];
  const float* be1  = (const float*)d_in[11];
  const float* g2   = (const float*)d_in[12];
  const float* be2  = (const float*)d_in[13];
  const float* W1   = (const float*)d_in[14];
  const float* c1   = (const float*)d_in[15];
  const float* W2   = (const float*)d_in[16];
  const float* c2   = (const float*)d_in[17];
  const float* Wfc  = (const float*)d_in[18];
  const float* bfc  = (const float*)d_in[19];
  float* out = (float*)d_out;

  // ---- workspace carve (256B aligned chunks)
  char* p = (char*)d_ws;
  auto carve = [&](size_t bytes) -> void* {
    void* r = (void*)p;
    p += (bytes + 255) & ~(size_t)255;
    return r;
  };
  float* h     = (float*)carve((size_t)Mm * Ee * 4);
  u16*   z     = (u16*)  carve((size_t)Mm * Ee * 2);
  u16*   qb    = (u16*)  carve((size_t)Mm * Aa * 2);
  u16*   kb    = (u16*)  carve((size_t)Mm * Aa * 2);
  u16*   vb    = (u16*)  carve((size_t)Mm * Aa * 2);
  u16*   headb = (u16*)  carve((size_t)Mm * Aa * 2);
  u16*   ffn1b = (u16*)  carve((size_t)Mm * Ff * 2);
  u16*   WqT   = (u16*)  carve((size_t)Aa * Ee * 2);
  u16*   WkT   = (u16*)  carve((size_t)Aa * Ee * 2);
  u16*   WvT   = (u16*)  carve((size_t)Aa * Ee * 2);
  u16*   WdrT  = (u16*)  carve((size_t)Ee * Aa * 2);
  u16*   W1T   = (u16*)  carve((size_t)Ff * Ee * 2);
  u16*   W2T   = (u16*)  carve((size_t)Ee * Ff * 2);
  u16*   WfcT  = (u16*)  carve((size_t)Vv * Ee * 2);
  (void)ws_size; (void)in_sizes; (void)n_in; (void)out_size;

  auto blocks = [](size_t total) { return (unsigned)((total + 255) / 256); };

  // ---- weight prep (bf16, [N,K] transposed for contiguous B-fragment pairs)
  transpose_to_bf16_kernel<<<blocks((size_t)Ee * Aa), 256, 0, stream>>>(Wq, WqT, Ee, Aa);
  transpose_to_bf16_kernel<<<blocks((size_t)Ee * Aa), 256, 0, stream>>>(Wk, WkT, Ee, Aa);
  transpose_to_bf16_kernel<<<blocks((size_t)Ee * Aa), 256, 0, stream>>>(Wv, WvT, Ee, Aa);
  wd_reduce_kernel<<<blocks((size_t)Ee * Aa), 256, 0, stream>>>(Wd, WdrT);
  transpose_to_bf16_kernel<<<blocks((size_t)Ee * Ff), 256, 0, stream>>>(W1, W1T, Ee, Ff);
  transpose_to_bf16_kernel<<<blocks((size_t)Ff * Ee), 256, 0, stream>>>(W2, W2T, Ff, Ee);
  transpose_to_bf16_kernel<<<blocks((size_t)Ee * Vv), 256, 0, stream>>>(Wfc, WfcT, Ee, Vv);

  // ---- embedding
  embed_kernel<<<blocks((size_t)Mm * Ee), 256, 0, stream>>>(x, emb, h);

  // ---- decoder stack (shared weights)
  for (int layer = 0; layer < STACKN; ++layer) {
    layernorm_kernel<<<Mm, 256, 0, stream>>>(h, g1, be1, z);
    gemm_bf16_kernel<<<dim3(Aa / GBN, Mm / GBM), 256, 0, stream>>>(
        z, WqT, bq, nullptr, nullptr, qb, Mm, Aa, Ee, 0);
    gemm_bf16_kernel<<<dim3(Aa / GBN, Mm / GBM), 256, 0, stream>>>(
        z, WkT, bk, nullptr, nullptr, kb, Mm, Aa, Ee, 0);
    gemm_bf16_kernel<<<dim3(Aa / GBN, Mm / GBM), 256, 0, stream>>>(
        z, WvT, bv, nullptr, nullptr, vb, Mm, Aa, Ee, 0);

    attention_kernel<<<(Bb * Ss / 16) / 4, 128, 0, stream>>>(qb, kb, vb, headb);

    // h += head @ Wd_red + bd   (16-head tile fold)
    gemm_bf16_kernel<<<dim3(Ee / GBN, Mm / GBM), 256, 0, stream>>>(
        headb, WdrT, bd, h, h, nullptr, Mm, Ee, Aa, 0);

    layernorm_kernel<<<Mm, 256, 0, stream>>>(h, g2, be2, z);
    gemm_bf16_kernel<<<dim3(Ff / GBN, Mm / GBM), 256, 0, stream>>>(
        z, W1T, c1, nullptr, nullptr, ffn1b, Mm, Ff, Ee, 1);
    gemm_bf16_kernel<<<dim3(Ee / GBN, Mm / GBM), 256, 0, stream>>>(
        ffn1b, W2T, c2, h, h, nullptr, Mm, Ee, Ff, 0);
  }

  // ---- logits: out = h @ Wfc + bfc  (reuse z as bf16(h))
  cvt_bf16_kernel<<<blocks((size_t)Mm * Ee), 256, 0, stream>>>(h, z, (size_t)Mm * Ee);
  gemm_bf16_kernel<<<dim3(Vv / GBN, Mm / GBM), 256, 0, stream>>>(
      z, WfcT, bfc, nullptr, out, nullptr, Mm, Vv, Ee, 0);
}
